// AfmoeSparseMoeBlock_38422777430201
// MI455X (gfx1250) — compile-verified, matrix operand order checked by reading, not run on hardware
//
#include <hip/hip_runtime.h>
#include <math.h>

#define Tn 1024
#define Hn 2048
#define Fn 1024
#define En 16
#define KP 80   // LDS row stride in halves (16B-aligned chunks, conflict-reducing)

typedef __attribute__((ext_vector_type(16))) _Float16     v16h;
typedef __attribute__((ext_vector_type(2)))  __fp16       fp16x2;
typedef __attribute__((ext_vector_type(8)))  float        v8f;
typedef __attribute__((ext_vector_type(4)))  float        f32x4;
typedef __attribute__((ext_vector_type(4)))  unsigned int u32x4;

union HFrag { u32x4 q[2]; v16h v; };

static __device__ inline unsigned pk2h(float a, float b) {
  union { fp16x2 h; unsigned u; } c;
  c.h = __builtin_amdgcn_cvt_pkrtz(a, b);   // v_cvt_pk_rtz_f16_f32
  return c.u;
}

static __device__ inline float siluf(float g) {
  return g / (1.0f + __expf(-g));
}

static __device__ inline v8f wmma16(const HFrag& a, const HFrag& b, v8f c) {
  return __builtin_amdgcn_wmma_f32_16x16x32_f16(false, a.v, false, b.v,
                                                (short)0, c, false, false);
}

// ---------------------------------------------------------------------------
// Router stage 1: scores[t][e] = sigmoid(x[t,:] @ gate_w[:,e]); one wave/token
// ---------------------------------------------------------------------------
__global__ __launch_bounds__(256) void router_logits_kernel(
    const float* __restrict__ x, const float* __restrict__ gw,
    float* __restrict__ scores) {
  const int lane = threadIdx.x & 31;
  const int wid  = threadIdx.x >> 5;
  const int tok  = blockIdx.x * 8 + wid;
  const float* xr = x + (size_t)tok * Hn;
  float acc[En];
#pragma unroll
  for (int e = 0; e < En; ++e) acc[e] = 0.0f;
  for (int h = lane; h < Hn; h += 32) {
    const float xv = xr[h];
    const float* g = gw + (size_t)h * En;
#pragma unroll
    for (int e = 0; e < En; ++e) acc[e] += xv * g[e];
  }
#pragma unroll
  for (int e = 0; e < En; ++e) {
    float v = acc[e];
    for (int m = 16; m > 0; m >>= 1) v += __shfl_xor(v, m, 32);
    acc[e] = v;
  }
  if (lane == 0) {
#pragma unroll
    for (int e = 0; e < En; ++e)
      scores[tok * En + e] = 1.0f / (1.0f + __expf(-acc[e]));
  }
}

// ---------------------------------------------------------------------------
// Router stage 2: top-2, weight normalization, per-expert compacted lists
// ---------------------------------------------------------------------------
__global__ __launch_bounds__(1024) void router_topk_kernel(
    const float* __restrict__ scores, const float* __restrict__ bias,
    int* __restrict__ counts_g, int* __restrict__ offs_g,
    int* __restrict__ tlist, float* __restrict__ tw) {
  __shared__ int cA[En];
  __shared__ int cB[En];
  __shared__ int oS[En];
  const int tok = threadIdx.x;
  if (tok < En) { cA[tok] = 0; cB[tok] = 0; }
  __syncthreads();

  float s[En];
#pragma unroll
  for (int e = 0; e < En; ++e) s[e] = scores[tok * En + e];

  int e0 = 0; float b0 = -1e30f;
#pragma unroll
  for (int e = 0; e < En; ++e) {
    float v = s[e] + bias[e];
    if (v > b0) { b0 = v; e0 = e; }
  }
  int e1 = 0; float b1 = -1e30f;
#pragma unroll
  for (int e = 0; e < En; ++e) {
    if (e == e0) continue;
    float v = s[e] + bias[e];
    if (v > b1) { b1 = v; e1 = e; }
  }
  float w0 = s[e0], w1 = s[e1];
  const float d = w0 + w1 + 1e-20f;
  w0 /= d; w1 /= d;                       // ROUTE_SCALE == 1.0

  atomicAdd(&cA[e0], 1);
  atomicAdd(&cA[e1], 1);
  __syncthreads();
  if (tok == 0) {
    int r = 0;
    for (int e = 0; e < En; ++e) { oS[e] = r; r += cA[e]; }
  }
  __syncthreads();
  const int p0 = oS[e0] + atomicAdd(&cB[e0], 1);
  tlist[p0] = tok; tw[p0] = w0;
  const int p1 = oS[e1] + atomicAdd(&cB[e1], 1);
  tlist[p1] = tok; tw[p1] = w1;
  if (tok < En) { counts_g[tok] = cA[tok]; offs_g[tok] = oS[tok]; }
}

// ---------------------------------------------------------------------------
// Shared expert gate+up: act_s = silu(x@Wg) * (x@Wu)
// 256 thr (8 waves, 2x4), tile 32(M)x64(N), K-step 64, register-pipelined.
// ---------------------------------------------------------------------------
__global__ __launch_bounds__(256) void sgu_kernel(
    const float* __restrict__ x, const float* __restrict__ wg,
    const float* __restrict__ wu, float* __restrict__ act) {
  __shared__ alignas(16) _Float16 As[32 * KP];
  __shared__ alignas(16) _Float16 Bg[64 * KP];
  __shared__ alignas(16) _Float16 Bu[64 * KP];
  const int tid = threadIdx.x, lane = tid & 31, wid = tid >> 5;
  const int waveM = wid >> 2, waveN = wid & 3;
  const int nb = blockIdx.x * 64, mb = blockIdx.y * 32;
  const int mA = tid >> 3, k8A = (tid & 7) * 8;
  const int kB = tid >> 4, n4B = (tid & 15) * 4;
  const int kq = (lane >> 4) * 8;
  const int arow = (waveM * 16 + (lane & 15)) * KP;
  const int brow = (waveN * 16 + (lane & 15)) * KP;
  v8f accG = {}, accU = {};
  f32x4 ra[2], rg[4], ru[4];

  {  // prologue: stage tile kb=0
    const float* ap = &x[(size_t)(mb + mA) * Hn + k8A];
    ra[0] = *(const f32x4*)ap; ra[1] = *(const f32x4*)(ap + 4);
#pragma unroll
    for (int j = 0; j < 4; ++j) {
      rg[j] = *(const f32x4*)&wg[(size_t)(kB + j * 16) * Fn + nb + n4B];
      ru[j] = *(const f32x4*)&wu[(size_t)(kB + j * 16) * Fn + nb + n4B];
    }
  }
  for (int kb = 0; kb < Hn; kb += 64) {
    __syncthreads();
    {  // commit staged registers to LDS (f32 -> f16)
      u32x4 pa;
      pa.x = pk2h(ra[0].x, ra[0].y); pa.y = pk2h(ra[0].z, ra[0].w);
      pa.z = pk2h(ra[1].x, ra[1].y); pa.w = pk2h(ra[1].z, ra[1].w);
      *(u32x4*)&As[mA * KP + k8A] = pa;
#pragma unroll
      for (int j = 0; j < 4; ++j) {
        const int k = kB + j * 16;
        Bg[(n4B + 0) * KP + k] = (_Float16)rg[j].x;
        Bg[(n4B + 1) * KP + k] = (_Float16)rg[j].y;
        Bg[(n4B + 2) * KP + k] = (_Float16)rg[j].z;
        Bg[(n4B + 3) * KP + k] = (_Float16)rg[j].w;
        Bu[(n4B + 0) * KP + k] = (_Float16)ru[j].x;
        Bu[(n4B + 1) * KP + k] = (_Float16)ru[j].y;
        Bu[(n4B + 2) * KP + k] = (_Float16)ru[j].z;
        Bu[(n4B + 3) * KP + k] = (_Float16)ru[j].w;
      }
    }
    if (kb + 64 < Hn) {  // issue next tile's loads; WMMAs below hide latency
      const float* ap = &x[(size_t)(mb + mA) * Hn + kb + 64 + k8A];
      ra[0] = *(const f32x4*)ap; ra[1] = *(const f32x4*)(ap + 4);
#pragma unroll
      for (int j = 0; j < 4; ++j) {
        rg[j] = *(const f32x4*)&wg[(size_t)(kb + 64 + kB + j * 16) * Fn + nb + n4B];
        ru[j] = *(const f32x4*)&wu[(size_t)(kb + 64 + kB + j * 16) * Fn + nb + n4B];
      }
    }
    __syncthreads();
#pragma unroll
    for (int g = 0; g < 2; ++g) {
      HFrag fa, fg, fu;
      const _Float16* ap = As + arow + g * 32 + kq;
      fa.q[0] = *(const u32x4*)ap; fa.q[1] = *(const u32x4*)(ap + 16);
      const _Float16* gp = Bg + brow + g * 32 + kq;
      fg.q[0] = *(const u32x4*)gp; fg.q[1] = *(const u32x4*)(gp + 16);
      const _Float16* up = Bu + brow + g * 32 + kq;
      fu.q[0] = *(const u32x4*)up; fu.q[1] = *(const u32x4*)(up + 16);
      accG = wmma16(fa, fg, accG);
      accU = wmma16(fa, fu, accU);
    }
  }

  const int mloc = 8 * (lane >> 4), nloc = lane & 15;
#pragma unroll
  for (int r = 0; r < 8; ++r) {
    const int row = mb + waveM * 16 + mloc + r;
    const int col = nb + waveN * 16 + nloc;
    act[(size_t)row * Fn + col] = siluf(accG[r]) * accU[r];
  }
}

// ---------------------------------------------------------------------------
// Shared expert down: out = act_s @ Wd (plain store; covers all T x H)
// ---------------------------------------------------------------------------
__global__ __launch_bounds__(256) void sdown_kernel(
    const float* __restrict__ act, const float* __restrict__ wd,
    float* __restrict__ out) {
  __shared__ alignas(16) _Float16 As[32 * KP];
  __shared__ alignas(16) _Float16 Bs[64 * KP];
  const int tid = threadIdx.x, lane = tid & 31, wid = tid >> 5;
  const int waveM = wid >> 2, waveN = wid & 3;
  const int nb = blockIdx.x * 64, mb = blockIdx.y * 32;
  const int mA = tid >> 3, k8A = (tid & 7) * 8;
  const int kB = tid >> 4, n4B = (tid & 15) * 4;
  const int kq = (lane >> 4) * 8;
  const int arow = (waveM * 16 + (lane & 15)) * KP;
  const int brow = (waveN * 16 + (lane & 15)) * KP;
  v8f acc = {};
  f32x4 ra[2], rb[4];

  {
    const float* ap = &act[(size_t)(mb + mA) * Fn + k8A];
    ra[0] = *(const f32x4*)ap; ra[1] = *(const f32x4*)(ap + 4);
#pragma unroll
    for (int j = 0; j < 4; ++j)
      rb[j] = *(const f32x4*)&wd[(size_t)(kB + j * 16) * Hn + nb + n4B];
  }
  for (int kb = 0; kb < Fn; kb += 64) {
    __syncthreads();
    {
      u32x4 pa;
      pa.x = pk2h(ra[0].x, ra[0].y); pa.y = pk2h(ra[0].z, ra[0].w);
      pa.z = pk2h(ra[1].x, ra[1].y); pa.w = pk2h(ra[1].z, ra[1].w);
      *(u32x4*)&As[mA * KP + k8A] = pa;
#pragma unroll
      for (int j = 0; j < 4; ++j) {
        const int k = kB + j * 16;
        Bs[(n4B + 0) * KP + k] = (_Float16)rb[j].x;
        Bs[(n4B + 1) * KP + k] = (_Float16)rb[j].y;
        Bs[(n4B + 2) * KP + k] = (_Float16)rb[j].z;
        Bs[(n4B + 3) * KP + k] = (_Float16)rb[j].w;
      }
    }
    if (kb + 64 < Fn) {
      const float* ap = &act[(size_t)(mb + mA) * Fn + kb + 64 + k8A];
      ra[0] = *(const f32x4*)ap; ra[1] = *(const f32x4*)(ap + 4);
#pragma unroll
      for (int j = 0; j < 4; ++j)
        rb[j] = *(const f32x4*)&wd[(size_t)(kb + 64 + kB + j * 16) * Hn + nb + n4B];
    }
    __syncthreads();
#pragma unroll
    for (int g = 0; g < 2; ++g) {
      HFrag fa, fb;
      const _Float16* ap = As + arow + g * 32 + kq;
      fa.q[0] = *(const u32x4*)ap; fa.q[1] = *(const u32x4*)(ap + 16);
      const _Float16* bp = Bs + brow + g * 32 + kq;
      fb.q[0] = *(const u32x4*)bp; fb.q[1] = *(const u32x4*)(bp + 16);
      acc = wmma16(fa, fb, acc);
    }
  }

  const int mloc = 8 * (lane >> 4), nloc = lane & 15;
#pragma unroll
  for (int r = 0; r < 8; ++r) {
    const int row = mb + waveM * 16 + mloc + r;
    const int col = nb + waveN * 16 + nloc;
    out[(size_t)row * Hn + col] = acc[r];
  }
}

// ---------------------------------------------------------------------------
// Routed experts gate+up: gathered A rows; silu*up epilogue -> act_r
// Grid: (Fn/64, Tn/32, En); blocks beyond count[e] exit early.
// ---------------------------------------------------------------------------
__global__ __launch_bounds__(256) void egu_kernel(
    const float* __restrict__ x, const float* __restrict__ wgu,
    const int* __restrict__ counts, const int* __restrict__ offs,
    const int* __restrict__ tlist, float* __restrict__ act) {
  const int e   = blockIdx.z;
  const int cnt = counts[e];
  const int mb  = blockIdx.y * 32;
  if (mb >= cnt) return;
  const int off = offs[e];
  const float* wbase = wgu + (size_t)e * Hn * (2 * Fn);

  __shared__ alignas(16) _Float16 As[32 * KP];
  __shared__ alignas(16) _Float16 Bg[64 * KP];
  __shared__ alignas(16) _Float16 Bu[64 * KP];
  const int tid = threadIdx.x, lane = tid & 31, wid = tid >> 5;
  const int waveM = wid >> 2, waveN = wid & 3;
  const int nb = blockIdx.x * 64;
  const int mA = tid >> 3, k8A = (tid & 7) * 8;
  const int kB = tid >> 4, n4B = (tid & 15) * 4;
  const int kq = (lane >> 4) * 8;
  const int arow = (waveM * 16 + (lane & 15)) * KP;
  const int brow = (waveN * 16 + (lane & 15)) * KP;
  const int slotA = mb + mA;
  const int tokA  = (slotA < cnt) ? tlist[off + slotA] : -1;
  v8f accG = {}, accU = {};
  f32x4 ra[2], rg[4], ru[4];

  {
    ra[0] = (f32x4){0.f, 0.f, 0.f, 0.f}; ra[1] = ra[0];
    if (tokA >= 0) {
      const float* ap = &x[(size_t)tokA * Hn + k8A];
      ra[0] = *(const f32x4*)ap; ra[1] = *(const f32x4*)(ap + 4);
    }
#pragma unroll
    for (int j = 0; j < 4; ++j) {
      const float* rowp = wbase + (size_t)(kB + j * 16) * (2 * Fn) + nb + n4B;
      rg[j] = *(const f32x4*)rowp;
      ru[j] = *(const f32x4*)(rowp + Fn);
    }
  }
  for (int kb = 0; kb < Hn; kb += 64) {
    __syncthreads();
    {
      u32x4 pa;
      pa.x = pk2h(ra[0].x, ra[0].y); pa.y = pk2h(ra[0].z, ra[0].w);
      pa.z = pk2h(ra[1].x, ra[1].y); pa.w = pk2h(ra[1].z, ra[1].w);
      *(u32x4*)&As[mA * KP + k8A] = pa;
#pragma unroll
      for (int j = 0; j < 4; ++j) {
        const int k = kB + j * 16;
        Bg[(n4B + 0) * KP + k] = (_Float16)rg[j].x;
        Bg[(n4B + 1) * KP + k] = (_Float16)rg[j].y;
        Bg[(n4B + 2) * KP + k] = (_Float16)rg[j].z;
        Bg[(n4B + 3) * KP + k] = (_Float16)rg[j].w;
        Bu[(n4B + 0) * KP + k] = (_Float16)ru[j].x;
        Bu[(n4B + 1) * KP + k] = (_Float16)ru[j].y;
        Bu[(n4B + 2) * KP + k] = (_Float16)ru[j].z;
        Bu[(n4B + 3) * KP + k] = (_Float16)ru[j].w;
      }
    }
    if (kb + 64 < Hn) {
      if (tokA >= 0) {
        const float* ap = &x[(size_t)tokA * Hn + kb + 64 + k8A];
        ra[0] = *(const f32x4*)ap; ra[1] = *(const f32x4*)(ap + 4);
      }
#pragma unroll
      for (int j = 0; j < 4; ++j) {
        const float* rowp =
            wbase + (size_t)(kb + 64 + kB + j * 16) * (2 * Fn) + nb + n4B;
        rg[j] = *(const f32x4*)rowp;
        ru[j] = *(const f32x4*)(rowp + Fn);
      }
    }
    __syncthreads();
#pragma unroll
    for (int g = 0; g < 2; ++g) {
      HFrag fa, fg, fu;
      const _Float16* ap = As + arow + g * 32 + kq;
      fa.q[0] = *(const u32x4*)ap; fa.q[1] = *(const u32x4*)(ap + 16);
      const _Float16* gp = Bg + brow + g * 32 + kq;
      fg.q[0] = *(const u32x4*)gp; fg.q[1] = *(const u32x4*)(gp + 16);
      const _Float16* up = Bu + brow + g * 32 + kq;
      fu.q[0] = *(const u32x4*)up; fu.q[1] = *(const u32x4*)(up + 16);
      accG = wmma16(fa, fg, accG);
      accU = wmma16(fa, fu, accU);
    }
  }

  const int mloc = 8 * (lane >> 4), nloc = lane & 15;
#pragma unroll
  for (int r = 0; r < 8; ++r) {
    const int slot = mb + waveM * 16 + mloc + r;
    if (slot < cnt) {
      const int col = nb + waveN * 16 + nloc;
      act[(size_t)(off + slot) * Fn + col] = siluf(accG[r]) * accU[r];
    }
  }
}

// ---------------------------------------------------------------------------
// Routed experts down: out[token,h] += tw * (act_r @ Wd_e)  (atomic scatter)
// Grid: (Hn/64, Tn/32, En)
// ---------------------------------------------------------------------------
__global__ __launch_bounds__(256) void edown_kernel(
    const float* __restrict__ act, const float* __restrict__ wde,
    const int* __restrict__ counts, const int* __restrict__ offs,
    const int* __restrict__ tlist, const float* __restrict__ tw,
    float* __restrict__ out) {
  const int e   = blockIdx.z;
  const int cnt = counts[e];
  const int mb  = blockIdx.y * 32;
  if (mb >= cnt) return;
  const int off = offs[e];
  const float* wbase = wde + (size_t)e * Fn * Hn;

  __shared__ alignas(16) _Float16 As[32 * KP];
  __shared__ alignas(16) _Float16 Bs[64 * KP];
  const int tid = threadIdx.x, lane = tid & 31, wid = tid >> 5;
  const int waveM = wid >> 2, waveN = wid & 3;
  const int nb = blockIdx.x * 64;
  const int mA = tid >> 3, k8A = (tid & 7) * 8;
  const int kB = tid >> 4, n4B = (tid & 15) * 4;
  const int kq = (lane >> 4) * 8;
  const int arow = (waveM * 16 + (lane & 15)) * KP;
  const int brow = (waveN * 16 + (lane & 15)) * KP;
  const int slotA = mb + mA;
  const bool okA = (slotA < cnt);
  v8f acc = {};
  f32x4 ra[2], rb[4];

  {
    ra[0] = (f32x4){0.f, 0.f, 0.f, 0.f}; ra[1] = ra[0];
    if (okA) {
      const float* ap = &act[(size_t)(off + slotA) * Fn + k8A];
      ra[0] = *(const f32x4*)ap; ra[1] = *(const f32x4*)(ap + 4);
    }
#pragma unroll
    for (int j = 0; j < 4; ++j)
      rb[j] = *(const f32x4*)&wbase[(size_t)(kB + j * 16) * Hn + nb + n4B];
  }
  for (int kb = 0; kb < Fn; kb += 64) {
    __syncthreads();
    {
      u32x4 pa;
      pa.x = pk2h(ra[0].x, ra[0].y); pa.y = pk2h(ra[0].z, ra[0].w);
      pa.z = pk2h(ra[1].x, ra[1].y); pa.w = pk2h(ra[1].z, ra[1].w);
      *(u32x4*)&As[mA * KP + k8A] = pa;
#pragma unroll
      for (int j = 0; j < 4; ++j) {
        const int k = kB + j * 16;
        Bs[(n4B + 0) * KP + k] = (_Float16)rb[j].x;
        Bs[(n4B + 1) * KP + k] = (_Float16)rb[j].y;
        Bs[(n4B + 2) * KP + k] = (_Float16)rb[j].z;
        Bs[(n4B + 3) * KP + k] = (_Float16)rb[j].w;
      }
    }
    if (kb + 64 < Fn) {
      if (okA) {
        const float* ap = &act[(size_t)(off + slotA) * Fn + kb + 64 + k8A];
        ra[0] = *(const f32x4*)ap; ra[1] = *(const f32x4*)(ap + 4);
      }
#pragma unroll
      for (int j = 0; j < 4; ++j)
        rb[j] = *(const f32x4*)&wbase[(size_t)(kb + 64 + kB + j * 16) * Hn + nb + n4B];
    }
    __syncthreads();
#pragma unroll
    for (int g = 0; g < 2; ++g) {
      HFrag fa, fb;
      const _Float16* ap = As + arow + g * 32 + kq;
      fa.q[0] = *(const u32x4*)ap; fa.q[1] = *(const u32x4*)(ap + 16);
      const _Float16* bp = Bs + brow + g * 32 + kq;
      fb.q[0] = *(const u32x4*)bp; fb.q[1] = *(const u32x4*)(bp + 16);
      acc = wmma16(fa, fb, acc);
    }
  }

  const int mloc = 8 * (lane >> 4), nloc = lane & 15;
#pragma unroll
  for (int r = 0; r < 8; ++r) {
    const int slot = mb + waveM * 16 + mloc + r;
    if (slot < cnt) {
      const int token = tlist[off + slot];
      const float wgt = tw[off + slot];
      const int col = nb + waveN * 16 + nloc;
      atomicAdd(&out[(size_t)token * Hn + col], wgt * acc[r]);
    }
  }
}

// ---------------------------------------------------------------------------
extern "C" void kernel_launch(void* const* d_in, const int* in_sizes, int n_in,
                              void* d_out, int out_size, void* d_ws, size_t ws_size,
                              hipStream_t stream) {
  (void)in_sizes; (void)n_in; (void)out_size; (void)ws_size;
  const float* x   = (const float*)d_in[0];  // [T, H]
  const float* gw  = (const float*)d_in[1];  // [H, E]
  const float* eb  = (const float*)d_in[2];  // [E]
  const float* sg  = (const float*)d_in[3];  // [H, F]
  const float* su  = (const float*)d_in[4];  // [H, F]
  const float* sd  = (const float*)d_in[5];  // [F, H]
  const float* egu = (const float*)d_in[6];  // [E, H, 2F]
  const float* edw = (const float*)d_in[7];  // [E, F, H]
  float* out = (float*)d_out;

  float* w      = (float*)d_ws;
  float* scores = w;                              // T*E
  int*   counts = (int*)(w + Tn * En);            // E
  int*   offs   = counts + En;                    // E
  int*   tlist  = offs + En;                      // 2*T
  float* tw     = (float*)(tlist + 2 * Tn);       // 2*T
  float* act_s  = tw + 2 * Tn;                    // T*F
  float* act_r  = act_s + (size_t)Tn * Fn;        // 2*T*F

  router_logits_kernel<<<Tn / 8, 256, 0, stream>>>(x, gw, scores);
  router_topk_kernel<<<1, Tn, 0, stream>>>(scores, eb, counts, offs, tlist, tw);
  sgu_kernel<<<dim3(Fn / 64, Tn / 32), 256, 0, stream>>>(x, sg, su, act_s);
  sdown_kernel<<<dim3(Hn / 64, Tn / 32), 256, 0, stream>>>(act_s, sd, out);
  egu_kernel<<<dim3(Fn / 64, Tn / 32, En), 256, 0, stream>>>(x, egu, counts, offs,
                                                             tlist, act_r);
  edown_kernel<<<dim3(Hn / 64, Tn / 32, En), 256, 0, stream>>>(act_r, edw, counts,
                                                               offs, tlist, tw, out);
}
